// HeadParallelDilatedAttention_42614665510999
// MI455X (gfx1250) — matrix operand order repaired and checked by reading, not compiled, not run on hardware
//
#include <hip/hip_runtime.h>
#include <hip/hip_bf16.h>
#include <math.h>

typedef __attribute__((ext_vector_type(16))) _Float16 v16h;
typedef __attribute__((ext_vector_type(8)))  float    v8f;

static __device__ __forceinline__ v8f wmma_f16(v16h a, v16h b, v8f c) {
    // D = A x B + C, 16x16x32, f16 inputs, f32 accumulate
    return __builtin_amdgcn_wmma_f32_16x16x32_f16(
        /*neg_a=*/false, a, /*neg_b=*/false, b,
        /*c_mod=*/(short)0, c, /*reuse_a=*/false, /*reuse_b=*/false);
}

// Problem constants (from reference setup_inputs)
#define NSEQ   4096
#define NHEAD  12
#define DHEAD  128
#define UNIQUE 1024                    // L/r == 1024 for all three groups
#define KLOG2E 0.12752401954424362f    // (1/sqrt(128)) * log2(e)

__global__ __launch_bounds__(256)
void dilated_flash_attn_kernel(const float* __restrict__ Q,
                               const float* __restrict__ K,
                               const float* __restrict__ V,
                               float* __restrict__ O,
                               const int* __restrict__ causal_flag)
{
    // K tile row-major [key][d], V tile transposed [d][key],
    // P conversion patch per wave (C-layout -> A-layout).
    __shared__ _Float16 sK[64][136];
    __shared__ _Float16 sVt[128][72];
    __shared__ _Float16 sP[8][16][72];

    const int tid   = threadIdx.x;
    const int lane  = tid & 31;
    const int wave  = tid >> 5;          // 8 waves, wave32
    const int qtile = blockIdx.x;        // 32 tiles of 128 q-rows
    const int head  = blockIdx.y;        // 12
    const int batch = blockIdx.z;        // 2

    const int group    = head >> 2;      // 4 heads per dilation group
    const int L        = 1024 << group;  // segment length
    const int r        = 1 << group;     // dilation rate
    const int qrow0    = qtile * 128;
    const int seg_base = qrow0 & ~(L - 1);
    const int q_in_seg0 = qrow0 - seg_base;

    // Causal mask only applies to segment 0 (reference semantics); otherwise
    // duplicated dilated keys cancel in softmax -> 1024 unique keys, stride r.
    const bool masked = (*causal_flag != 0) && (seg_base == 0);
    const int  kend   = masked ? ((L < q_in_seg0 + 128) ? L : (q_in_seg0 + 128))
                               : UNIQUE;

    // ---- Q (16 rows per wave) into WMMA A fragments ---------------------
    // A layout (16-bit 16x32): row = lane%16; lanes<16 hold K {0-7,16-23},
    // lanes>=16 hold K {8-15,24-31}.
    const int qrow = qrow0 + wave * 16 + (lane & 15);
    const float* qptr = Q + ((size_t)((batch * NSEQ + qrow) * NHEAD + head)) * DHEAD;
    const int akb = (lane & 16) ? 8 : 0;
    v16h qa[4];
#pragma unroll
    for (int c = 0; c < 4; ++c) {
#pragma unroll
        for (int i = 0; i < 8; ++i) {
            qa[c][i]     = (_Float16)qptr[32 * c + akb + i];
            qa[c][i + 8] = (_Float16)qptr[32 * c + akb + 16 + i];
        }
    }

    // ---- online-softmax state ------------------------------------------
    v8f acc[8];
#pragma unroll
    for (int n = 0; n < 8; ++n) {
#pragma unroll
        for (int vv = 0; vv < 8; ++vv) acc[n][vv] = 0.0f;
    }
    float mrow[8], lrow[8];
#pragma unroll
    for (int vv = 0; vv < 8; ++vv) { mrow[vv] = -3.0e38f; lrow[vv] = 0.0f; }

    const int keyA  = lane & 15;
    const int kdHalf = (lane & 16) ? 16 : 0;
    const int rbC   = (lane & 16) ? 8 : 0;     // C-layout row base

    // ---- main loop: 64 keys per tile -----------------------------------
    for (int kt = 0; kt < kend; kt += 64) {
        __syncthreads();   // previous tile fully consumed

        // Cooperative load: K row-major f16, V transposed f16.
        // Dilation gather: key position in segment = ((kt+j) & 1023) * r.
#pragma unroll
        for (int it = 0; it < 8; ++it) {
            const int flat = tid + 256 * it;   // 0..2047
            const int j    = flat >> 5;        // key row 0..63
            const int c4   = flat & 31;        // float4 column
            const int kvpos = seg_base + (((kt + j) & (UNIQUE - 1)) * r);
            const size_t base = ((size_t)((batch * NSEQ + kvpos) * NHEAD + head)) * DHEAD + c4 * 4;
            const float4 k4 = *reinterpret_cast<const float4*>(K + base);
            const float4 v4 = *reinterpret_cast<const float4*>(V + base);
            sK[j][c4 * 4 + 0] = (_Float16)k4.x;
            sK[j][c4 * 4 + 1] = (_Float16)k4.y;
            sK[j][c4 * 4 + 2] = (_Float16)k4.z;
            sK[j][c4 * 4 + 3] = (_Float16)k4.w;
            sVt[c4 * 4 + 0][j] = (_Float16)v4.x;
            sVt[c4 * 4 + 1][j] = (_Float16)v4.y;
            sVt[c4 * 4 + 2][j] = (_Float16)v4.z;
            sVt[c4 * 4 + 3][j] = (_Float16)v4.w;
        }
        __syncthreads();

        // ---- S = Q * K^T : 4 fragments of 16x16 (keys 16n..16n+15) ------
        // B layout (16-bit 32x16): col = lane%16; lanes<16 hold K {0-7,16-23},
        // lanes>=16 hold K {8-15,24-31} (K = head-dim chunk).
        v8f s[4];
#pragma unroll
        for (int n = 0; n < 4; ++n)
#pragma unroll
            for (int vv = 0; vv < 8; ++vv) s[n][vv] = 0.0f;
#pragma unroll
        for (int c = 0; c < 4; ++c) {
#pragma unroll
            for (int n = 0; n < 4; ++n) {
                v16h b;
#pragma unroll
                for (int i = 0; i < 16; ++i)
                    b[i] = sK[16 * n + keyA][32 * c + (kdHalf >> 1) + (i & 7) + ((i & 8) << 1)];
                s[n] = wmma_f16(qa[c], b, s[n]);
            }
        }

        // ---- causal mask (segment 0 only), dilated-position based -------
        if (masked) {
#pragma unroll
            for (int n = 0; n < 4; ++n) {
                const int km = kt + 16 * n + (lane & 15);
#pragma unroll
                for (int vv = 0; vv < 8; ++vv) {
                    const int qr = q_in_seg0 + wave * 16 + vv + rbC;
                    if (km > qr) s[n][vv] = -1.0e30f;
                }
            }
        }

        // ---- online softmax --------------------------------------------
        float p[4][8], rs[8], resc[8];
        bool upd = false;
#pragma unroll
        for (int vv = 0; vv < 8; ++vv) {
            float t = fmaxf(fmaxf(s[0][vv], s[1][vv]), fmaxf(s[2][vv], s[3][vv]));
            t = fmaxf(t, __shfl_xor(t, 1));
            t = fmaxf(t, __shfl_xor(t, 2));
            t = fmaxf(t, __shfl_xor(t, 4));
            t = fmaxf(t, __shfl_xor(t, 8));
            const float mn = fmaxf(mrow[vv], t);
            upd = upd || (mn > mrow[vv]);
            resc[vv] = exp2f((mrow[vv] - mn) * KLOG2E);
            mrow[vv] = mn;
#pragma unroll
            for (int n = 0; n < 4; ++n)
                p[n][vv] = exp2f((s[n][vv] - mn) * KLOG2E);
            float r2 = (p[0][vv] + p[1][vv]) + (p[2][vv] + p[3][vv]);
            r2 += __shfl_xor(r2, 1);
            r2 += __shfl_xor(r2, 2);
            r2 += __shfl_xor(r2, 4);
            r2 += __shfl_xor(r2, 8);
            rs[vv] = r2;
        }
        // rescale only when some row-max actually advanced (wave-uniform)
        if (__any(upd ? 1 : 0)) {
#pragma unroll
            for (int vv = 0; vv < 8; ++vv) {
                lrow[vv] *= resc[vv];
#pragma unroll
                for (int n = 0; n < 8; ++n) acc[n][vv] *= resc[vv];
            }
        }
#pragma unroll
        for (int vv = 0; vv < 8; ++vv) lrow[vv] += rs[vv];

        // ---- stage P: C layout -> LDS -> A layout (wave-private) --------
        {
            const int col0 = lane & 15;
#pragma unroll
            for (int vv = 0; vv < 8; ++vv)
#pragma unroll
                for (int n = 0; n < 4; ++n)
                    sP[wave][vv + rbC][16 * n + col0] = (_Float16)p[n][vv];
        }
        // cross-lane LDS RAW within one wave: DS ops are per-wave in-order;
        // force the wait and forbid compiler reordering.
        asm volatile("s_wait_dscnt 0" ::: "memory");

        v16h pa0, pa1;
        {
            const int prow = lane & 15;
            const int pkb  = (lane & 16) ? 8 : 0;
#pragma unroll
            for (int i = 0; i < 8; ++i) {
                pa0[i]     = sP[wave][prow][pkb + i];
                pa0[i + 8] = sP[wave][prow][pkb + 16 + i];
                pa1[i]     = sP[wave][prow][32 + pkb + i];
                pa1[i + 8] = sP[wave][prow][32 + pkb + 16 + i];
            }
        }

        // ---- O += P * V (8 col tiles x two k=32 halves) -----------------
        const int vcol = lane & 15;
        const int vkh  = (lane & 16) ? 16 : 0;
#pragma unroll
        for (int n = 0; n < 8; ++n) {
            v16h vb0, vb1;
#pragma unroll
            for (int i = 0; i < 16; ++i) {
                vb0[i] = sVt[16 * n + vcol][(vkh >> 1) + (i & 7) + ((i & 8) << 1)];
                vb1[i] = sVt[16 * n + vcol][32 + (vkh >> 1) + (i & 7) + ((i & 8) << 1)];
            }
            acc[n] = wmma_f16(pa0, vb0, acc[n]);
            acc[n] = wmma_f16(pa1, vb1, acc[n]);
        }
    }

    // ---- normalize and store (f32, (b, n, h, d) layout) -----------------
    const int colL = lane & 15;
#pragma unroll
    for (int vv = 0; vv < 8; ++vv) {
        const float inv = 1.0f / lrow[vv];
        const int pos = qrow0 + wave * 16 + vv + rbC;
        float* op = O + ((size_t)((batch * NSEQ + pos) * NHEAD + head)) * DHEAD;
#pragma unroll
        for (int n = 0; n < 8; ++n) op[16 * n + colL] = acc[n][vv] * inv;
    }
}

extern "C" void kernel_launch(void* const* d_in, const int* in_sizes, int n_in,
                              void* d_out, int out_size, void* d_ws, size_t ws_size,
                              hipStream_t stream) {
    (void)in_sizes; (void)n_in; (void)out_size; (void)d_ws; (void)ws_size;
    const float* q = (const float*)d_in[0];
    const float* k = (const float*)d_in[1];
    const float* v = (const float*)d_in[2];
    const int* causal = (const int*)d_in[3];
    float* out = (float*)d_out;

    dim3 grid(NSEQ / 128, NHEAD, 2);   // (q-tile, head, batch)
    dilated_flash_attn_kernel<<<grid, 256, 0, stream>>>(q, k, v, out, causal);
}
